// MultiHeadAttention_28157805592698
// MI455X (gfx1250) — matrix operand order repaired and checked by reading, not compiled, not run on hardware
//
#include <hip/hip_runtime.h>
#include <hip/hip_bf16.h>

// ---------------------------------------------------------------------------
// MHA for B=4, S=1024, d_model=256, H=8, head_dim=256 on gfx1250 (MI455X).
// bf16 WMMA (v_wmma_f32_16x16x32_bf16), f32 accumulate, flash-attention
// fused softmax, async LDS staging (global_load_async_to_lds_b128) of K.
// ---------------------------------------------------------------------------

typedef __attribute__((ext_vector_type(16))) __bf16 v16bf;
typedef __attribute__((ext_vector_type(8)))  float  v8f;

#define BATCH 4
#define SEQ   1024
#define DMODEL 256
#define NHEADS 8
#define HDIM  256                 // head_dim == d_model here
#define DPROJ (NHEADS * HDIM)     // 2048
#define MROWS (BATCH * SEQ)       // 4096

#define KSTRIDE 40                // padded LDS row stride for K tile (bf16)
#define PSTRIDE 40                // padded LDS row stride for P tile (bf16)

__device__ __forceinline__ unsigned short f2bf(float f) {
  unsigned int u = __float_as_uint(f);
  u += 0x7FFFu + ((u >> 16) & 1u);            // round-to-nearest-even
  return (unsigned short)(u >> 16);
}

__device__ __forceinline__ v8f wmma_bf16(v16bf a, v16bf b, v8f c) {
  return __builtin_amdgcn_wmma_f32_16x16x32_bf16(false, a, false, b,
                                                 (short)0, c, false, false);
}

// async 16B copy: global (saddr + byte_off) -> LDS (lds byte address)
__device__ __forceinline__ void async_ld_b128(void* lds_dst,
                                              const unsigned short* gbase,
                                              int byte_off) {
  unsigned lds = (unsigned)(unsigned long long)lds_dst;   // low 32b = LDS addr
  asm volatile("global_load_async_to_lds_b128 %0, %1, %2 offset:0"
               :: "v"(lds), "v"(byte_off),
                  "s"((unsigned long long)gbase)
               : "memory");
}

// --------------------------- fp32 -> bf16 convert --------------------------
__global__ void k_f32_to_bf16(const float* __restrict__ in,
                              unsigned short* __restrict__ out, int n) {
  int i = blockIdx.x * 256 + threadIdx.x;
  if (i < n) out[i] = f2bf(in[i]);
}

// ------------------------ fused QKV projection GEMM ------------------------
// grid (MROWS/32, DPROJ/64, 3), block 32: one wave computes a 32x64 tile
// (2x4 register blocking -> 8 WMMAs per 6 fragment loads).
// z=0: q -> [B,H,S,256]; z=1: k -> kT [B,H,256,S]; z=2: v -> [B,H,S,256]
__global__ void __launch_bounds__(32) k_qkv_proj(
    const unsigned short* __restrict__ Qb, const unsigned short* __restrict__ Kb,
    const unsigned short* __restrict__ Vb,
    const unsigned short* __restrict__ Wqb, const unsigned short* __restrict__ Wkb,
    const unsigned short* __restrict__ Wvb,
    const float* __restrict__ bq, const float* __restrict__ bk,
    const float* __restrict__ bv,
    unsigned short* __restrict__ qp, unsigned short* __restrict__ kTp,
    unsigned short* __restrict__ vp) {
  const int lane = threadIdx.x, col = lane & 15, hi = lane >> 4;
  const int m0 = blockIdx.x * 32, n0 = blockIdx.y * 64, z = blockIdx.z;

  const unsigned short* X = (z == 0) ? Qb : (z == 1) ? Kb : Vb;
  const unsigned short* W = (z == 0) ? Wqb : (z == 1) ? Wkb : Wvb;
  const float* bias = (z == 0) ? bq : (z == 1) ? bk : bv;

  v8f acc[2][4];
#pragma unroll
  for (int i = 0; i < 2; i++)
#pragma unroll
    for (int j = 0; j < 4; j++) acc[i][j] = (v8f){};

#pragma unroll
  for (int k0 = 0; k0 < DMODEL; k0 += 32) {
    v16bf a[2], b[4];
#pragma unroll
    for (int i = 0; i < 2; i++)
      a[i] = *(const v16bf*)&X[(size_t)(m0 + i * 16 + col) * DMODEL + k0 + hi * 16];
#pragma unroll
    for (int j = 0; j < 4; j++)
      for (int s = 0; s < 16; s++)
        b[j][s] = __builtin_bit_cast(__bf16, W[(size_t)(k0 + hi * 16 + s) * DPROJ + n0 + j * 16 + col]);
#pragma unroll
    for (int i = 0; i < 2; i++)
#pragma unroll
      for (int j = 0; j < 4; j++) acc[i][j] = wmma_bf16(a[i], b[j], acc[i][j]);
  }

#pragma unroll
  for (int j = 0; j < 4; j++) {
    const int nb = n0 + j * 16;
    const float bval = bias[nb + col];
    const int h = nb >> 8;
    const int dd = (nb & 255) + col;
#pragma unroll
    for (int i = 0; i < 2; i++) {
#pragma unroll
      for (int r = 0; r < 8; r++) {
        const int m = m0 + i * 16 + r + 8 * hi;   // global row = b*S + s
        const int b_ = m >> 10, s = m & 1023;
        const unsigned short val = f2bf(acc[i][j][r] + bval);
        const size_t head = (size_t)(b_ * NHEADS + h);
        if (z == 1) {
          kTp[(head * HDIM + dd) * SEQ + s] = val;        // K transposed
        } else {
          unsigned short* dst = (z == 0) ? qp : vp;
          dst[(head * SEQ + s) * HDIM + dd] = val;
        }
      }
    }
  }
}

// --------------------------- flash attention -------------------------------
// grid (SEQ/64, B*H), block 128 (4 waves; wave w owns q-rows m0+w*16..+15).
// The 256x32 K^T tile is shared by all 4 waves: double-buffered in LDS via
// global_load_async_to_lds_b128, overlapped with compute (s_wait_asynccnt 8).
__global__ void __launch_bounds__(128) k_flash_attn(
    const unsigned short* __restrict__ qp, const unsigned short* __restrict__ kTp,
    const unsigned short* __restrict__ vp, unsigned short* __restrict__ attn_out) {
  __shared__ unsigned short Ktile[2][HDIM * KSTRIDE];  // 2 x 20 KB
  __shared__ unsigned short pbuf[4][16 * PSTRIDE];     // 5 KB

  const int t = threadIdx.x;
  const int lane = t & 31, wid = t >> 5;
  const int col = lane & 15, hi = (lane >> 4) & 1;
  const int m0 = blockIdx.x * 64 + wid * 16;  // this wave's q-tile start
  const int bh = blockIdx.y;                  // b*H + h

  const unsigned short* Qh = qp  + (size_t)bh * SEQ * HDIM;
  const unsigned short* Kh = kTp + (size_t)bh * HDIM * SEQ;
  const unsigned short* Vh = vp  + (size_t)bh * SEQ * HDIM;
  unsigned short* pb = pbuf[wid];

  // K-tile async copy: 256 rows x 64B, 128 threads -> 2 rows, 8 issues each
  auto issue_ktile = [&](int j0, int buf) {
#pragma unroll
    for (int rr = 0; rr < 2; rr++) {
      const int row = t * 2 + rr;
#pragma unroll
      for (int q = 0; q < 4; q++)
        async_ld_b128(&Ktile[buf][row * KSTRIDE + q * 8], Kh,
                      (row * SEQ + j0 + q * 8) * 2);
    }
  };

  // Q fragments for all 8 k-chunks of head_dim, resident in VGPRs
  v16bf qf[8];
#pragma unroll
  for (int kc = 0; kc < 8; kc++)
    qf[kc] = *(const v16bf*)&Qh[(size_t)(m0 + col) * HDIM + kc * 32 + hi * 16];

  v8f acc[16];
#pragma unroll
  for (int tt = 0; tt < 16; tt++) acc[tt] = (v8f){};
  float mrow[8], lrow[8];
#pragma unroll
  for (int r = 0; r < 8; r++) { mrow[r] = -1e30f; lrow[r] = 0.0f; }

  issue_ktile(0, 0);                           // prologue: tile 0 in flight

  for (int j0 = 0; j0 < SEQ; j0 += 32) {
    const int cur = (j0 >> 5) & 1;
    if (j0 + 32 < SEQ) {
      issue_ktile(j0 + 32, cur ^ 1);           // prefetch next tile
      asm volatile("s_wait_asynccnt 8" ::: "memory");   // current tile done
    } else {
      asm volatile("s_wait_asynccnt 0" ::: "memory");
    }
    __syncthreads();                           // tile visible to all waves
    const unsigned short* Kt = Ktile[cur];

    // --- scores: two 16x16 tiles from LDS K fragments ---
    v8f s0 = {}, s1 = {};
#pragma unroll
    for (int kc = 0; kc < 8; kc++) {
      v16bf kf;
      for (int s = 0; s < 16; s++) kf[s] = __builtin_bit_cast(__bf16, Kt[(kc * 32 + hi * 16 + s) * KSTRIDE + col]);
      s0 = wmma_bf16(qf[kc], kf, s0);
    }
#pragma unroll
    for (int kc = 0; kc < 8; kc++) {
      v16bf kf;
      for (int s = 0; s < 16; s++) kf[s] = __builtin_bit_cast(__bf16, Kt[(kc * 32 + hi * 16 + s) * KSTRIDE + 16 + col]);
      s1 = wmma_bf16(qf[kc], kf, s1);
    }

    // --- online softmax update (scale = 1/sqrt(256) = 1/16) ---
#pragma unroll
    for (int r = 0; r < 8; r++) {
      float a0 = s0[r] * 0.0625f;
      float a1 = s1[r] * 0.0625f;
      float mx = fmaxf(a0, a1);
#pragma unroll
      for (int off = 8; off; off >>= 1) mx = fmaxf(mx, __shfl_xor(mx, off, 16));
      const float mn = fmaxf(mrow[r], mx);
      const float sc = __expf(mrow[r] - mn);
      const float p0 = __expf(a0 - mn);
      const float p1 = __expf(a1 - mn);
      float rs = p0 + p1;
#pragma unroll
      for (int off = 8; off; off >>= 1) rs += __shfl_xor(rs, off, 16);
      lrow[r] = lrow[r] * sc + rs;
      mrow[r] = mn;
#pragma unroll
      for (int tt = 0; tt < 16; tt++) acc[tt][r] *= sc;
      const int row = r + 8 * hi;              // C-layout row of this lane
      pb[row * PSTRIDE + col]      = f2bf(p0); // bf16 P tile via LDS
      pb[row * PSTRIDE + 16 + col] = f2bf(p1);
    }

    // --- reload P in A-fragment layout, multiply by V (global, coalesced) ---
    v16bf pf = *(const v16bf*)&pb[col * PSTRIDE + hi * 16];
#pragma unroll
    for (int tt = 0; tt < 16; tt++) {
      v16bf vf;
      for (int s = 0; s < 16; s++) vf[s] = __builtin_bit_cast(__bf16, Vh[(size_t)(j0 + hi * 16 + s) * HDIM + tt * 16 + col]);
      acc[tt] = wmma_bf16(pf, vf, acc[tt]);
    }
    __syncthreads();                           // done reading before overwrite
  }

  // --- epilogue: normalize, write [B,S,H*256] bf16 ---
  const int b_ = bh >> 3, h = bh & 7;
#pragma unroll
  for (int r = 0; r < 8; r++) {
    const float inv = 1.0f / lrow[r];
    const int s = m0 + r + 8 * hi;
    const size_t rowb = ((size_t)(b_ * SEQ + s)) * DPROJ + h * HDIM;
#pragma unroll
    for (int tt = 0; tt < 16; tt++)
      attn_out[rowb + tt * 16 + col] = f2bf(acc[tt][r] * inv);
  }
}

// --------------------------- output projection -----------------------------
// grid (MROWS/32, DMODEL/64), block 32: 32x64 tile per wave, 2x4 blocking.
__global__ void __launch_bounds__(32) k_out_proj(
    const unsigned short* __restrict__ A, const unsigned short* __restrict__ Wob,
    const float* __restrict__ bo, float* __restrict__ out) {
  const int lane = threadIdx.x, col = lane & 15, hi = lane >> 4;
  const int m0 = blockIdx.x * 32, n0 = blockIdx.y * 64;
  v8f acc[2][4];
#pragma unroll
  for (int i = 0; i < 2; i++)
#pragma unroll
    for (int j = 0; j < 4; j++) acc[i][j] = (v8f){};

#pragma unroll 4
  for (int k0 = 0; k0 < DPROJ; k0 += 32) {
    v16bf a[2], b[4];
#pragma unroll
    for (int i = 0; i < 2; i++)
      a[i] = *(const v16bf*)&A[(size_t)(m0 + i * 16 + col) * DPROJ + k0 + hi * 16];
#pragma unroll
    for (int j = 0; j < 4; j++)
      for (int s = 0; s < 16; s++)
        b[j][s] = __builtin_bit_cast(__bf16, Wob[(size_t)(k0 + hi * 16 + s) * DMODEL + n0 + j * 16 + col]);
#pragma unroll
    for (int i = 0; i < 2; i++)
#pragma unroll
      for (int j = 0; j < 4; j++) acc[i][j] = wmma_bf16(a[i], b[j], acc[i][j]);
  }

#pragma unroll
  for (int j = 0; j < 4; j++) {
    const float bval = bo[n0 + j * 16 + col];
#pragma unroll
    for (int i = 0; i < 2; i++)
#pragma unroll
      for (int r = 0; r < 8; r++)
        out[(size_t)(m0 + i * 16 + r + 8 * hi) * DMODEL + n0 + j * 16 + col] =
            acc[i][j][r] + bval;
  }
}

// ---------------------------------------------------------------------------
extern "C" void kernel_launch(void* const* d_in, const int* in_sizes, int n_in,
                              void* d_out, int out_size, void* d_ws, size_t ws_size,
                              hipStream_t stream) {
  (void)in_sizes; (void)n_in; (void)out_size; (void)ws_size;
  const float* Q  = (const float*)d_in[0];
  const float* K  = (const float*)d_in[1];
  const float* V  = (const float*)d_in[2];
  const float* Wq = (const float*)d_in[3];
  const float* bq = (const float*)d_in[4];
  const float* Wk = (const float*)d_in[5];
  const float* bk = (const float*)d_in[6];
  const float* Wv = (const float*)d_in[7];
  const float* bv = (const float*)d_in[8];
  const float* Wo = (const float*)d_in[9];
  const float* bo = (const float*)d_in[10];

  char* ws = (char*)d_ws;
  size_t off = 0;
  auto carve = [&](size_t elems) -> unsigned short* {
    unsigned short* p = (unsigned short*)(ws + off);
    off += (elems * sizeof(unsigned short) + 255) & ~(size_t)255;
    return p;
  };
  unsigned short* Qb  = carve((size_t)MROWS * DMODEL);
  unsigned short* Kb  = carve((size_t)MROWS * DMODEL);
  unsigned short* Vb  = carve((size_t)MROWS * DMODEL);
  unsigned short* Wqb = carve((size_t)DMODEL * DPROJ);
  unsigned short* Wkb = carve((size_t)DMODEL * DPROJ);
  unsigned short* Wvb = carve((size_t)DMODEL * DPROJ);
  unsigned short* Wob = carve((size_t)DPROJ * DMODEL);
  unsigned short* qp  = carve((size_t)BATCH * NHEADS * SEQ * HDIM);
  unsigned short* kTp = carve((size_t)BATCH * NHEADS * HDIM * SEQ);
  unsigned short* vp  = carve((size_t)BATCH * NHEADS * SEQ * HDIM);
  unsigned short* ao  = carve((size_t)MROWS * DPROJ);

  auto conv = [&](const float* in, unsigned short* outp, int n) {
    k_f32_to_bf16<<<(n + 255) / 256, 256, 0, stream>>>(in, outp, n);
  };
  conv(Q,  Qb,  MROWS * DMODEL);
  conv(K,  Kb,  MROWS * DMODEL);
  conv(V,  Vb,  MROWS * DMODEL);
  conv(Wq, Wqb, DMODEL * DPROJ);
  conv(Wk, Wkb, DMODEL * DPROJ);
  conv(Wv, Wvb, DMODEL * DPROJ);
  conv(Wo, Wob, DPROJ * DMODEL);

  k_qkv_proj<<<dim3(MROWS / 32, DPROJ / 64, 3), 32, 0, stream>>>(
      Qb, Kb, Vb, Wqb, Wkb, Wvb, bq, bk, bv, qp, kTp, vp);

  k_flash_attn<<<dim3(SEQ / 64, BATCH * NHEADS), 128, 0, stream>>>(
      qp, kTp, vp, ao);

  k_out_proj<<<dim3(MROWS / 32, DMODEL / 64), 32, 0, stream>>>(
      ao, Wob, bo, (float*)d_out);
}
